// QKVAttention_14516989460772
// MI455X (gfx1250) — matrix operand order repaired and checked by reading, not compiled, not run on hardware
//
#include <hip/hip_runtime.h>

// ---------------------------------------------------------------------------
// Flash-attention, qkv (4, 3*8*64, 2048) fp32, head dim 64, T=2048.
// Pre-pass: f16 copies of K (transposed [s][c], scaled) and V ([c][s]) in d_ws.
// Main: 1 workgroup = 1 head x 64 query rows, 128 thr = 4 waves.
//   K/V tiles streamed by the Tensor Data Mover (tensor_load_to_lds),
//   double-buffered, padded rows (72 halves) via TDM pad feature.
//   S = QK^T and O += P V^T via v_wmma_f32_16x16x32_f16 (16 per iteration).
//   Online softmax row reductions via DPP row_ror (pure VALU, no LDS).
// ---------------------------------------------------------------------------

typedef __attribute__((ext_vector_type(16))) _Float16 v16h;
typedef __attribute__((ext_vector_type(8)))  _Float16 v8h;
typedef __attribute__((ext_vector_type(8)))  float    v8f;
typedef __attribute__((ext_vector_type(4)))  unsigned int v4u;
typedef __attribute__((ext_vector_type(8)))  int v8i;
typedef __attribute__((ext_vector_type(4)))  int v4i;

#define T_LEN 2048
#define CH    64
#define BM    64            // query rows per workgroup
#define BN    64            // key cols per inner iteration
#define NW    4
#define NTHR  128
#define NIT   (T_LEN / BN)  // 32

#define QT_S  72            // padded LDS row strides (halves)
#define KV_S  72            // 64 data + 8 pad, produced by TDM padding
#define P_S   72
#define O_S   66            // floats

#define KT_ELEMS (32ull * T_LEN * CH)   // f16 elems of Kt block in ws

// ---- DPP row_ror reductions within each 16-lane row -----------------------
#define ROR_STEP_MAX(x, N)                                                    \
  x = fmaxf(x, __int_as_float(__builtin_amdgcn_update_dpp(                    \
          __float_as_int(x), __float_as_int(x), 0x120 + (N), 0xf, 0xf, false)))
#define ROR_STEP_ADD(x, N)                                                    \
  x = x + __int_as_float(__builtin_amdgcn_update_dpp(                         \
          __float_as_int(x), __float_as_int(x), 0x120 + (N), 0xf, 0xf, false))

__device__ __forceinline__ float rowmax16(float x) {
  ROR_STEP_MAX(x, 1); ROR_STEP_MAX(x, 2); ROR_STEP_MAX(x, 4); ROR_STEP_MAX(x, 8);
  return x;
}
__device__ __forceinline__ float rowsum16(float x) {
  ROR_STEP_ADD(x, 1); ROR_STEP_ADD(x, 2); ROR_STEP_ADD(x, 4); ROR_STEP_ADD(x, 8);
  return x;
}

// ---- TDM: 2D f16 tile -> LDS, rows padded 64->72 halves -------------------
__device__ __forceinline__ unsigned lds_off(const void* p) {
  return (unsigned)(unsigned long long)(uintptr_t)p;   // low 32 bits = LDS addr
}

__device__ __forceinline__ void tdm_load_tile_f16(
    unsigned lds_addr, unsigned long long gaddr,
    unsigned tile_d0 /*elems per row*/, unsigned tile_d1 /*rows*/,
    unsigned long long stride0 /*elems between rows*/) {
  v4u g0;
  g0[0] = 1u;                                        // count=1 (valid user D#)
  g0[1] = lds_addr;                                  // lds_addr
  g0[2] = (unsigned)(gaddr & 0xffffffffu);           // global_addr[31:0]
  g0[3] = ((unsigned)(gaddr >> 32) & 0x01ffffffu)    // global_addr[56:32]
        | (2u << 30);                                // type = 2 (image)
  v8i g1;
  g1[0] = (int)((1u << 16)      // data_size = 1 -> 2-byte elements
              | (1u << 20)      // pad_enable
              | (4u << 22)      // pad_interval: 32 dwords (one 64-half row)
              | (3u << 25));    // pad_amount: 4 dwords (8 halves) -> stride 72
  g1[1] = (int)((tile_d0 & 0xffffu) << 16);                       // tensor_dim0 lo
  g1[2] = (int)(((tile_d0 >> 16) & 0xffffu)
              | ((tile_d1 & 0xffffu) << 16));                     // d0 hi | tensor_dim1 lo
  g1[3] = (int)(((tile_d1 >> 16) & 0xffffu)
              | ((tile_d0 & 0xffffu) << 16));                     // d1 hi | tile_dim0
  g1[4] = (int)(tile_d1 & 0xffffu);                               // tile_dim1 (tile_dim2=0)
  g1[5] = (int)(stride0 & 0xffffffffu);                           // tensor_dim0_stride lo
  g1[6] = (int)((stride0 >> 32) & 0xffffu);                       // stride0 hi (stride1=0)
  g1[7] = 0;
  v4i z4; z4[0] = 0; z4[1] = 0; z4[2] = 0; z4[3] = 0;
  v8i z8; z8[0] = 0; z8[1] = 0; z8[2] = 0; z8[3] = 0;
  z8[4] = 0; z8[5] = 0; z8[6] = 0; z8[7] = 0;
  // 6-arg toolchain form: (g0, g1, g2, g3, g4, cpol)
  __builtin_amdgcn_tensor_load_to_lds(g0, g1, z4, z4, z8, 0);
}

// ---------------------------------------------------------------------------
// Pre-pass: build f16 Kt[bh][s][c] (scaled) and Vc[bh][c][s] in workspace.
// ---------------------------------------------------------------------------
__global__ __launch_bounds__(256)
void qkv_prep_kernel(const float* __restrict__ qkv, _Float16* __restrict__ ws) {
  __shared__ _Float16 sT[64 * 72];
  const int tid = threadIdx.x;
  const int bh  = blockIdx.y;
  const int b   = bh >> 3, hh = bh & 7;
  const int s0  = blockIdx.x * 64;
  const size_t bk = ((size_t)b * 1536 + 512  + hh * 64) * T_LEN;
  const size_t bv = ((size_t)b * 1536 + 1024 + hh * 64) * T_LEN;
  const float scale = 0.3535533905932738f;
  _Float16* Kt = ws + (size_t)bh * T_LEN * CH;
  _Float16* Vc = ws + KT_ELEMS + (size_t)bh * CH * T_LEN;
#pragma unroll
  for (int i = 0; i < 16; ++i) {
    int e = i * 256 + tid;
    int c = e >> 6, s = e & 63;
    sT[c * 72 + s] = (_Float16)(qkv[bk + (size_t)c * T_LEN + (s0 + s)] * scale);
    Vc[(size_t)c * T_LEN + (s0 + s)] =
        (_Float16)qkv[bv + (size_t)c * T_LEN + (s0 + s)];
  }
  __syncthreads();
#pragma unroll
  for (int i = 0; i < 16; ++i) {
    int e = i * 256 + tid;
    int s = e >> 6, c = e & 63;
    Kt[(size_t)(s0 + s) * CH + c] = sT[c * 72 + s];
  }
}

// ---------------------------------------------------------------------------
// Main flash-attention kernel.
// ---------------------------------------------------------------------------
__global__ __launch_bounds__(NTHR)
void qkv_attn_fa_kernel(const float* __restrict__ qkv,
                        const _Float16* __restrict__ ws,
                        float* __restrict__ out) {
  __shared__ __align__(16) _Float16 sQT[BM * QT_S];
  __shared__ __align__(16) _Float16 sK [2][BN * KV_S];
  __shared__ __align__(16) _Float16 sVb[2][CH * KV_S];
  __shared__ __align__(16) _Float16 sP [NW * 16 * P_S];
  __shared__ __align__(16) float    sO [CH * O_S];

  const int tid  = threadIdx.x;
  const int wave = tid >> 5;
  const int lane = tid & 31;
  const int ln   = lane & 15;
  const int hf   = lane >> 4;

  const int bh = blockIdx.y;
  const int b  = bh >> 3;
  const int hh = bh & 7;
  const size_t bq = ((size_t)b * 1536 + hh * 64) * T_LEN;
  const size_t bo = ((size_t)b * 512  + hh * 64) * T_LEN;
  const int t0 = blockIdx.x * BM;

  const _Float16* Kt = ws + (size_t)bh * T_LEN * CH;            // [s][c] f16
  const _Float16* Vc = ws + KT_ELEMS + (size_t)bh * CH * T_LEN; // [c][s] f16

  const float scale = 0.3535533905932738f;

  // kick off TDM for tile 0 while we stage Q
  if (wave == 0) {
    tdm_load_tile_f16(lds_off(&sK[0][0]),
                      (unsigned long long)(uintptr_t)(Kt + 0 * CH),
                      CH, BN, CH);
    tdm_load_tile_f16(lds_off(&sVb[0][0]),
                      (unsigned long long)(uintptr_t)(Vc + 0),
                      BN, CH, T_LEN);
  }

  // ---- stage Q tile transposed [t][c], fp16, pre-scaled ------------------
#pragma unroll
  for (int i = 0; i < (BM * CH) / NTHR; ++i) {
    int e = i * NTHR + tid;
    int c = e >> 6, t = e & 63;
    float v = qkv[bq + (size_t)c * T_LEN + (t0 + t)] * scale;
    sQT[t * QT_S + c] = (_Float16)v;
  }
  __syncthreads();

  // ---- Q A-fragments (K-dim 64 -> two 16x16x32 steps) --------------------
  v16h aq[2];
#pragma unroll
  for (int kk = 0; kk < 2; ++kk) {
    int c1 = kk * 32 + hf * 8;
    v8h lo = *(const v8h*)&sQT[ln * QT_S + c1];
    v8h hi = *(const v8h*)&sQT[ln * QT_S + c1 + 16];
    v16h a;
#pragma unroll
    for (int i = 0; i < 8; ++i) { a[i] = lo[i]; a[i + 8] = hi[i]; }
    aq[kk] = a;
  }

  v8f oacc[4];
#pragma unroll
  for (int cf = 0; cf < 4; ++cf) oacc[cf] = (v8f)0.0f;
  float mrow[8], lrow[8];
#pragma unroll
  for (int r = 0; r < 8; ++r) { mrow[r] = -__builtin_inff(); lrow[r] = 0.0f; }

  for (int it = 0; it < NIT; ++it) {
    const int cur = it & 1;
    if (wave == 0) {
      if (it + 1 < NIT) {   // stream next tile into the other buffer
        const int sn = (it + 1) * BN;
        tdm_load_tile_f16(lds_off(&sK[cur ^ 1][0]),
                          (unsigned long long)(uintptr_t)(Kt + (size_t)sn * CH),
                          CH, BN, CH);
        tdm_load_tile_f16(lds_off(&sVb[cur ^ 1][0]),
                          (unsigned long long)(uintptr_t)(Vc + sn),
                          BN, CH, T_LEN);
        __builtin_amdgcn_s_wait_tensorcnt(2);   // current tile (older 2) done
      } else {
        __builtin_amdgcn_s_wait_tensorcnt(0);
      }
    }
    __syncthreads();
    const _Float16* kb = sK[cur];
    const _Float16* vb = sVb[cur];

    // ---- S = (Qs)(Ks)^T : four 16-column subtiles ------------------------
    v8f sfr[4];
#pragma unroll
    for (int j = 0; j < 4; ++j) {
      v8f acc = (v8f)0.0f;
#pragma unroll
      for (int kk = 0; kk < 2; ++kk) {
        v8h lo = *(const v8h*)&kb[(j * 16 + ln) * KV_S + kk * 32 + hf * 8];
        v8h hi = *(const v8h*)&kb[(j * 16 + ln) * KV_S + kk * 32 + hf * 8 + 16];
        v16h bfr;
#pragma unroll
        for (int i = 0; i < 8; ++i) { bfr[i] = lo[i]; bfr[i + 8] = hi[i]; }
        acc = __builtin_amdgcn_wmma_f32_16x16x32_f16(
            false, aq[kk], false, bfr, (short)0, acc, false, false);
      }
      sfr[j] = acc;
    }

    // ---- online softmax (DPP row reductions, no LDS) ---------------------
#pragma unroll
    for (int r = 0; r < 8; ++r) {
      float v = fmaxf(fmaxf(sfr[0][r], sfr[1][r]), fmaxf(sfr[2][r], sfr[3][r]));
      v = rowmax16(v);
      float mn    = fmaxf(mrow[r], v);
      float alpha = __expf(mrow[r] - mn);
      float p0 = __expf(sfr[0][r] - mn);
      float p1 = __expf(sfr[1][r] - mn);
      float p2 = __expf(sfr[2][r] - mn);
      float p3 = __expf(sfr[3][r] - mn);
      float rs = rowsum16((p0 + p1) + (p2 + p3));
      lrow[r] = lrow[r] * alpha + rs;
      mrow[r] = mn;
#pragma unroll
      for (int cf = 0; cf < 4; ++cf) oacc[cf][r] *= alpha;
      _Float16* pp = &sP[(wave * 16 + (r + 8 * hf)) * P_S];
      pp[ 0 + ln] = (_Float16)p0;
      pp[16 + ln] = (_Float16)p1;
      pp[32 + ln] = (_Float16)p2;
      pp[48 + ln] = (_Float16)p3;
    }

    // ---- O += P V^T : K-dim 64 -> two k-steps ----------------------------
#pragma unroll
    for (int kk = 0; kk < 2; ++kk) {
      v16h ap;
      {
        const _Float16* base = &sP[(wave * 16 + ln) * P_S + kk * 32];
        v8h lo = *(const v8h*)&base[hf * 8];
        v8h hi = *(const v8h*)&base[hf * 8 + 16];
#pragma unroll
        for (int i = 0; i < 8; ++i) { ap[i] = lo[i]; ap[i + 8] = hi[i]; }
      }
#pragma unroll
      for (int cf = 0; cf < 4; ++cf) {
        v8h lo = *(const v8h*)&vb[(cf * 16 + ln) * KV_S + kk * 32 + hf * 8];
        v8h hi = *(const v8h*)&vb[(cf * 16 + ln) * KV_S + kk * 32 + hf * 8 + 16];
        v16h bfr;
#pragma unroll
        for (int i = 0; i < 8; ++i) { bfr[i] = lo[i]; bfr[i + 8] = hi[i]; }
        oacc[cf] = __builtin_amdgcn_wmma_f32_16x16x32_f16(
            false, ap, false, bfr, (short)0, oacc[cf], false, false);
      }
    }
    __syncthreads();   // everyone done with buf[cur] before it is refilled
  }

  // ---- epilogue: normalize, transpose via LDS, coalesced store -----------
#pragma unroll
  for (int cf = 0; cf < 4; ++cf) {
#pragma unroll
    for (int r = 0; r < 8; ++r) {
      float inv = 1.0f / lrow[r];
      sO[(cf * 16 + ln) * O_S + (wave * 16 + r + 8 * hf)] = oacc[cf][r] * inv;
    }
  }
  __syncthreads();
#pragma unroll
  for (int i = 0; i < (BM * CH) / NTHR; ++i) {
    int e = i * NTHR + tid;
    int c = e >> 6, t = e & 63;
    out[bo + (size_t)c * T_LEN + (t0 + t)] = sO[c * O_S + t];
  }
}

extern "C" void kernel_launch(void* const* d_in, const int* in_sizes, int n_in,
                              void* d_out, int out_size, void* d_ws, size_t ws_size,
                              hipStream_t stream) {
  const float* qkv = (const float*)d_in[0];
  float* out = (float*)d_out;
  _Float16* ws = (_Float16*)d_ws;   // needs 16 MB (2 x 8 MB f16 K/V copies)

  dim3 pgrid(T_LEN / 64, 32);
  qkv_prep_kernel<<<pgrid, 256, 0, stream>>>(qkv, ws);

  dim3 grid(T_LEN / BM, 32);
  qkv_attn_fa_kernel<<<grid, NTHR, 0, stream>>>(qkv, ws, out);
}
